// Attention_88948772700322
// MI455X (gfx1250) — matrix operand order repaired and checked by reading, not compiled, not run on hardware
//
#include <hip/hip_runtime.h>

typedef __attribute__((ext_vector_type(16))) _Float16 v16h;
typedef __attribute__((ext_vector_type(8)))  float    v8f;

#define DEV static __device__ __forceinline__

DEV v8f wmma16(v16h a, v16h b, v8f c) {
  return __builtin_amdgcn_wmma_f32_16x16x32_f16(false, a, false, b, (short)0, c, false, false);
}

// A fragment (16x32 f16, M x K): lane = l + 16*ph ; row = row0+l
// ph=0: K = k0+0..7 and k0+16..23 ; ph=1: K = k0+8..15 and k0+24..31
DEV v16h a_frag_f32(const float* __restrict__ base, int ld, int row0, int k0, int l, int ph) {
  const float* p = base + (size_t)(row0 + l) * ld + k0 + ph * 8;
  v16h a;
#pragma unroll
  for (int i = 0; i < 8; ++i) a[i] = (_Float16)p[i];
#pragma unroll
  for (int i = 0; i < 8; ++i) a[8 + i] = (_Float16)p[16 + i];
  return a;
}
DEV v16h a_frag_f16(const _Float16* base, int ld, int row0, int k0, int l, int ph) {
  const _Float16* p = base + (size_t)(row0 + l) * ld + k0 + ph * 8;
  v16h a;
#pragma unroll
  for (int i = 0; i < 8; ++i) a[i] = p[i];
#pragma unroll
  for (int i = 0; i < 8; ++i) a[8 + i] = p[16 + i];
  return a;
}
// B fragment (32x16 f16, K x N): lane column n = col0+l, same K order as A (ph=0: K 0..7,16..23; ph=1: K 8..15,24..31).
// source row (col0+l) of `base` holds the K run (i.e. stored K-contiguous per column).
DEV v16h b_frag_f32(const float* __restrict__ base, int ld, int col0, int k0, int l, int ph) {
  const float* p = base + (size_t)(col0 + l) * ld + k0 + ph * 8;
  v16h b;
#pragma unroll
  for (int i = 0; i < 8; ++i) { b[i] = (_Float16)p[i]; b[8 + i] = (_Float16)p[16 + i]; }
  return b;
}
DEV v16h b_frag_f16(const _Float16* base, int ld, int col0, int k0, int l, int ph) {
  const _Float16* p = base + (size_t)(col0 + l) * ld + k0 + ph * 8;
  v16h b;
#pragma unroll
  for (int i = 0; i < 8; ++i) { b[i] = p[i]; b[8 + i] = p[16 + i]; }
  return b;
}

DEV float half_reduce_max(float v) {
#pragma unroll
  for (int m = 8; m >= 1; m >>= 1) v = fmaxf(v, __shfl_xor(v, m, 32));
  return v;
}
DEV float half_reduce_sum(float v) {
#pragma unroll
  for (int m = 8; m >= 1; m >>= 1) v += __shfl_xor(v, m, 32);
  return v;
}

// ---------------- Kernel 1: QKV projection -------------------------------
// x: [4096 rows][1024], wqkv: [3072][1024] ; out: q,k [B,H,N,64] f16 ; vT [B,H,64,N] f16
__global__ __launch_bounds__(256) void qkv_kernel(const float* __restrict__ x,
                                                  const float* __restrict__ wqkv,
                                                  _Float16* __restrict__ qb,
                                                  _Float16* __restrict__ kb,
                                                  _Float16* __restrict__ vtb) {
  const int tid = threadIdx.x, lane = tid & 31, wid = tid >> 5;
  const int l = lane & 15, ph = lane >> 4;
  const int row0 = blockIdx.y * 128 + (wid >> 2) * 64;
  const int col0 = blockIdx.x * 256 + (wid & 3) * 64;

  v8f acc[4][4] = {};
  for (int k0 = 0; k0 < 1024; k0 += 32) {
    v16h af[4], bf[4];
#pragma unroll
    for (int mi = 0; mi < 4; ++mi) af[mi] = a_frag_f32(x, 1024, row0 + mi * 16, k0, l, ph);
#pragma unroll
    for (int ni = 0; ni < 4; ++ni) bf[ni] = b_frag_f32(wqkv, 1024, col0 + ni * 16, k0, l, ph);
#pragma unroll
    for (int mi = 0; mi < 4; ++mi)
#pragma unroll
      for (int ni = 0; ni < 4; ++ni) acc[mi][ni] = wmma16(af[mi], bf[ni], acc[mi][ni]);
  }
  // scatter into q/k/vT layouts
#pragma unroll
  for (int mi = 0; mi < 4; ++mi)
#pragma unroll
    for (int ni = 0; ni < 4; ++ni) {
      v8f c = acc[mi][ni];
      int ncol = col0 + ni * 16 + l;            // 0..3071
      int which = ncol >> 10;                   // 0:q 1:k 2:v
      int cd = ncol & 1023;
      int hh = cd >> 6, e = cd & 63;
#pragma unroll
      for (int i = 0; i < 8; ++i) {
        int r = row0 + mi * 16 + i + 8 * ph;    // row = n*4 + b
        int n = r >> 2, b = r & 3;
        _Float16 val = (_Float16)c[i];
        if (which == 0)
          qb[((size_t)((b * 16 + hh) * 1024 + n)) * 64 + e] = val;
        else if (which == 1)
          kb[((size_t)((b * 16 + hh) * 1024 + n)) * 64 + e] = val;
        else
          vtb[((size_t)((b * 16 + hh) * 64 + e)) * 1024 + n] = val;
      }
    }
}

// ---------------- Kernel 2: flash attention ------------------------------
// grid: (8 row-blocks, 64 bh) ; block 256 = 8 waves, each wave owns 16 query rows.
// K and V(transposed) 64x64 tiles are staged once per workgroup in LDS and shared
// by all 8 waves; next tile is prefetched (global_prefetch_b8) during compute.
__global__ __launch_bounds__(256) void attn_kernel(const _Float16* __restrict__ qb,
                                                   const _Float16* __restrict__ kb,
                                                   const _Float16* __restrict__ vtb,
                                                   _Float16* __restrict__ ao) {
  __shared__ _Float16 ktile[64 * 64];           // [keyrow j][d]      8 KB
  __shared__ _Float16 vtile[64 * 64];           // [e][keyrow j]      8 KB
  __shared__ _Float16 pbuf[8 * 16 * 64];        // per-wave P staging 16 KB
  const int tid = threadIdx.x, lane = tid & 31, wid = tid >> 5;
  const int l = lane & 15, ph = lane >> 4;
  const int bh = blockIdx.y, b = bh >> 4, hh = bh & 15;
  const int nq0 = blockIdx.x * 128 + wid * 16;

  const _Float16* qbase = qb + (size_t)bh * 1024 * 64;
  const _Float16* kbase = kb + (size_t)bh * 1024 * 64;
  const _Float16* vbase = vtb + (size_t)bh * 64 * 1024;
  _Float16* pw = pbuf + wid * 16 * 64;

  // cooperative-fill indices: each of 256 threads owns a 16-half (32B) chunk
  const int fr = tid >> 2;                      // row 0..63
  const int fc = (tid & 3) * 16;                // chunk 0/16/32/48

  v16h qa[2];
#pragma unroll
  for (int s = 0; s < 2; ++s) qa[s] = a_frag_f16(qbase, 64, nq0, s * 32, l, ph);

  float ms[8], ls[8];
  v8f oacc[4] = {};
#pragma unroll
  for (int i = 0; i < 8; ++i) { ms[i] = -3.0e38f; ls[i] = 0.f; }

  for (int j0 = 0; j0 < 1024; j0 += 64) {
    // stage K tile [j][d] and transposed V tile [e][j] in LDS
    *(v16h*)&ktile[fr * 64 + fc] = *(const v16h*)&kbase[(size_t)(j0 + fr) * 64 + fc];
    *(v16h*)&vtile[fr * 64 + fc] = *(const v16h*)&vbase[(size_t)fr * 1024 + j0 + fc];
    __syncthreads();
    // prefetch next tile into the cache hierarchy while we compute this one
    if (j0 + 64 < 1024) {
      __builtin_prefetch(&kbase[(size_t)(j0 + 64 + fr) * 64 + fc], 0, 1);
      __builtin_prefetch(&vbase[(size_t)fr * 1024 + j0 + 64 + fc], 0, 1);
    }

    v8f sacc[4] = {};
#pragma unroll
    for (int s = 0; s < 2; ++s)
#pragma unroll
      for (int ni = 0; ni < 4; ++ni) {
        v16h kf = b_frag_f16(ktile, 64, ni * 16, s * 32, l, ph);
        sacc[ni] = wmma16(qa[s], kf, sacc[ni]);
      }
    // online softmax (scale = hd^-0.5 = 0.125)
#pragma unroll
    for (int i = 0; i < 8; ++i) {
      float mloc = -3.0e38f;
#pragma unroll
      for (int ni = 0; ni < 4; ++ni) {
        sacc[ni][i] *= 0.125f;
        mloc = fmaxf(mloc, sacc[ni][i]);
      }
      mloc = half_reduce_max(mloc);
      float mnew = fmaxf(ms[i], mloc);
      float alpha = __expf(ms[i] - mnew);
      float rsum = 0.f;
#pragma unroll
      for (int ni = 0; ni < 4; ++ni) {
        float p = __expf(sacc[ni][i] - mnew);
        sacc[ni][i] = p;
        rsum += p;
      }
      rsum = half_reduce_sum(rsum);
      ls[i] = ls[i] * alpha + rsum;
      ms[i] = mnew;
#pragma unroll
      for (int ni = 0; ni < 4; ++ni) oacc[ni][i] *= alpha;
    }
    // C-layout -> A-layout round trip through this wave's LDS region
#pragma unroll
    for (int ni = 0; ni < 4; ++ni)
#pragma unroll
      for (int i = 0; i < 8; ++i)
        pw[(i + 8 * ph) * 64 + ni * 16 + l] = (_Float16)sacc[ni][i];
    asm volatile("s_wait_dscnt 0" ::: "memory");  // cross-lane LDS visibility (intra-wave)
#pragma unroll
    for (int s = 0; s < 2; ++s) {
      v16h pa = a_frag_f16(pw, 64, 0, s * 32, l, ph);
#pragma unroll
      for (int ni = 0; ni < 4; ++ni) {
        v16h vf = b_frag_f16(vtile, 64, ni * 16, s * 32, l, ph);
        oacc[ni] = wmma16(pa, vf, oacc[ni]);
      }
    }
    __syncthreads();  // tiles consumed; safe to refill next iteration
  }
  // normalize and store in the reference's permuted layout: ao[n][h*256 + b*64 + e]
#pragma unroll
  for (int ni = 0; ni < 4; ++ni)
#pragma unroll
    for (int i = 0; i < 8; ++i) {
      int n = nq0 + i + 8 * ph;
      int e = ni * 16 + l;
      ao[(size_t)n * 4096 + hh * 256 + b * 64 + e] = (_Float16)(oacc[ni][i] / ls[i]);
    }
}

// ---------------- Kernel 3: output projection + bias ---------------------
// ao acts as [4096 rows][1024] row-major; out[row][d] = ao_row . wproj[d,:] + bias[d]
__global__ __launch_bounds__(256) void proj_kernel(const _Float16* __restrict__ ao,
                                                   const float* __restrict__ wproj,
                                                   const float* __restrict__ bias,
                                                   float* __restrict__ out) {
  const int tid = threadIdx.x, lane = tid & 31, wid = tid >> 5;
  const int l = lane & 15, ph = lane >> 4;
  const int row0 = blockIdx.y * 128 + (wid >> 2) * 64;
  const int col0 = blockIdx.x * 256 + (wid & 3) * 64;

  v8f acc[4][4] = {};
  for (int k0 = 0; k0 < 1024; k0 += 32) {
    v16h af[4], bf[4];
#pragma unroll
    for (int mi = 0; mi < 4; ++mi) af[mi] = a_frag_f16(ao, 1024, row0 + mi * 16, k0, l, ph);
#pragma unroll
    for (int ni = 0; ni < 4; ++ni) bf[ni] = b_frag_f32(wproj, 1024, col0 + ni * 16, k0, l, ph);
#pragma unroll
    for (int mi = 0; mi < 4; ++mi)
#pragma unroll
      for (int ni = 0; ni < 4; ++ni) acc[mi][ni] = wmma16(af[mi], bf[ni], acc[mi][ni]);
  }
#pragma unroll
  for (int mi = 0; mi < 4; ++mi)
#pragma unroll
    for (int ni = 0; ni < 4; ++ni) {
      int col = col0 + ni * 16 + l;
      float bv = bias[col];
#pragma unroll
      for (int i = 0; i < 8; ++i) {
        int r = row0 + mi * 16 + i + 8 * ph;
        out[(size_t)r * 1024 + col] = acc[mi][ni][i] + bv;
      }
    }
}

extern "C" void kernel_launch(void* const* d_in, const int* in_sizes, int n_in,
                              void* d_out, int out_size, void* d_ws, size_t ws_size,
                              hipStream_t stream) {
  (void)in_sizes; (void)n_in; (void)out_size; (void)ws_size;
  const float* x      = (const float*)d_in[0];  // [1024,4,1024]
  const float* w_qkv  = (const float*)d_in[1];  // [3072,1024]
  const float* w_proj = (const float*)d_in[2];  // [1024,1024]
  const float* b_proj = (const float*)d_in[3];  // [1024]
  float* out = (float*)d_out;

  char* ws = (char*)d_ws;
  const size_t QKV = (size_t)4 * 16 * 1024 * 64 * sizeof(_Float16);  // 8 MB each
  _Float16* qb  = (_Float16*)(ws);
  _Float16* kb  = (_Float16*)(ws + QKV);
  _Float16* vtb = (_Float16*)(ws + 2 * QKV);
  _Float16* ao  = (_Float16*)(ws + 3 * QKV);    // [1024][4096] f16, 8 MB

  qkv_kernel <<<dim3(12, 32), dim3(256), 0, stream>>>(x, w_qkv, qb, kb, vtb);
  attn_kernel<<<dim3(8, 64),  dim3(256), 0, stream>>>(qb, kb, vtb, ao);
  proj_kernel<<<dim3(4, 32),  dim3(256), 0, stream>>>(ao, w_proj, b_proj, out);
}
